// QuantLinear_31044023615637
// MI455X (gfx1250) — matrix operand order repaired and checked by reading, not compiled, not run on hardware
//
#include <hip/hip_runtime.h>

typedef __attribute__((ext_vector_type(16))) _Float16 v16h;
typedef __attribute__((ext_vector_type(8)))  _Float16 v8h;
typedef __attribute__((ext_vector_type(8)))  float    v8f;
typedef __attribute__((ext_vector_type(4)))  float    f4;
typedef __attribute__((ext_vector_type(4)))  int      i4;

#define TM 128
#define TN 128
#define TK 64
#define LDK (TK + 8)   // padded stride in halves (144 B) to spread LDS banks

__global__ __launch_bounds__(256, 1)
void QuantLinear_wmma_kernel(const float* __restrict__ X,
                             const int*   __restrict__ Wq,
                             const float* __restrict__ scale,
                             const float* __restrict__ bias,
                             float* __restrict__ Y,
                             int M, int N, int K)
{
    // double-buffered f16 tiles: 2 * (18432 + 18432) B = 72 KB LDS
    __shared__ __align__(16) _Float16 As[2][TM * LDK];
    __shared__ __align__(16) _Float16 Bs[2][TN * LDK];

    const int tid  = threadIdx.x;
    const int lane = tid & 31;
    const int wave = tid >> 5;
    const int wm   = wave >> 1;      // 0..3  (M direction, 32 rows each)
    const int wn   = wave & 1;       // 0..1  (N direction, 64 cols each)

    const int bm = blockIdx.y * TM;
    const int bn = blockIdx.x * TN;

    const int g   = lane >> 4;       // lane group (0: lanes 0-15, 1: lanes 16-31)
    const int l16 = lane & 15;

    v8f acc[2][4];
#pragma unroll
    for (int i = 0; i < 2; ++i)
#pragma unroll
        for (int j = 0; j < 4; ++j)
            acc[i][j] = (v8f){0.f,0.f,0.f,0.f,0.f,0.f,0.f,0.f};

    // staging decomposition: each thread owns 8 contiguous k per row,
    // 4 rows strided by 32 -> 128x64 tile with 256 threads
    const int sr = tid >> 3;           // base row 0..31
    const int sc = (tid & 7) << 3;     // column (elements): 0,8,...,56

    // in-flight K-tile held in registers (software pipeline stage)
    f4 ar[8];
    i4 br[8];

    // ---- prologue: issue loads for K-tile 0 ----
#pragma unroll
    for (int i = 0; i < 4; ++i) {
        const int r = sr + i * 32;
        ar[2*i+0] = *(const f4*)(X  + (size_t)(bm + r) * K + sc);
        ar[2*i+1] = *(const f4*)(X  + (size_t)(bm + r) * K + sc + 4);
        br[2*i+0] = *(const i4*)(Wq + (size_t)(bn + r) * K + sc);
        br[2*i+1] = *(const i4*)(Wq + (size_t)(bn + r) * K + sc + 4);
    }

    int cur = 0;
    for (int kt = 0; kt < K; kt += TK) {
        // ---- drain in-flight loads: convert f32/i16 -> f16, b128 LDS stores ----
#pragma unroll
        for (int i = 0; i < 4; ++i) {
            const int r = sr + i * 32;
            v8h ha;
            ha[0] = (_Float16)ar[2*i+0].x; ha[1] = (_Float16)ar[2*i+0].y;
            ha[2] = (_Float16)ar[2*i+0].z; ha[3] = (_Float16)ar[2*i+0].w;
            ha[4] = (_Float16)ar[2*i+1].x; ha[5] = (_Float16)ar[2*i+1].y;
            ha[6] = (_Float16)ar[2*i+1].z; ha[7] = (_Float16)ar[2*i+1].w;
            *(v8h*)&As[cur][r * LDK + sc] = ha;
            v8h hb;   // i32 values fit i16 exactly -> direct i16->f16 convert
            hb[0] = (_Float16)(short)br[2*i+0].x; hb[1] = (_Float16)(short)br[2*i+0].y;
            hb[2] = (_Float16)(short)br[2*i+0].z; hb[3] = (_Float16)(short)br[2*i+0].w;
            hb[4] = (_Float16)(short)br[2*i+1].x; hb[5] = (_Float16)(short)br[2*i+1].y;
            hb[6] = (_Float16)(short)br[2*i+1].z; hb[7] = (_Float16)(short)br[2*i+1].w;
            *(v8h*)&Bs[cur][r * LDK + sc] = hb;
        }
        __syncthreads();

        // ---- issue global loads for the NEXT K-tile (overlap with WMMAs) ----
        if (kt + TK < K) {
            const int kn = kt + TK;
#pragma unroll
            for (int i = 0; i < 4; ++i) {
                const int r = sr + i * 32;
                ar[2*i+0] = *(const f4*)(X  + (size_t)(bm + r) * K + kn + sc);
                ar[2*i+1] = *(const f4*)(X  + (size_t)(bm + r) * K + kn + sc + 4);
                br[2*i+0] = *(const i4*)(Wq + (size_t)(bn + r) * K + kn + sc);
                br[2*i+1] = *(const i4*)(Wq + (size_t)(bn + r) * K + kn + sc + 4);
            }
        }

        // ---- two 16x16x32 k-steps over the staged K-tile ----
#pragma unroll
        for (int ks = 0; ks < TK; ks += 32) {
            // A fragments: lane holds row m; halves 0-7 -> k = ks+8g+0..7,
            //              halves 8-15 -> k = ks+8g+16..23  (ISA 16-bit A layout)
            v16h a[2];
#pragma unroll
            for (int i = 0; i < 2; ++i) {
                const int m = wm * 32 + i * 16 + l16;
                v8h lo = *(const v8h*)&As[cur][m * LDK + ks + 8 * g];
                v8h hi = *(const v8h*)&As[cur][m * LDK + ks + 8 * g + 16];
                a[i] = __builtin_shufflevector(lo, hi,
                        0,1,2,3,4,5,6,7,8,9,10,11,12,13,14,15);
            }
            // B fragments: lane holds col n; halves 0-7 -> k = ks+8g+0..7, halves 8-15 -> k = ks+8g+16..23
            v16h b[4];
#pragma unroll
            for (int j = 0; j < 4; ++j) {
                const int n = wn * 64 + j * 16 + l16;
                v8h lo = *(const v8h*)&Bs[cur][n * LDK + ks + 8 * g];
                v8h hi = *(const v8h*)&Bs[cur][n * LDK + ks + 8 * g + 16];
                b[j] = __builtin_shufflevector(lo, hi,
                        0,1,2,3,4,5,6,7,8,9,10,11,12,13,14,15);
            }
#pragma unroll
            for (int i = 0; i < 2; ++i)
#pragma unroll
                for (int j = 0; j < 4; ++j)
                    acc[i][j] = __builtin_amdgcn_wmma_f32_16x16x32_f16(
                        false, a[i], false, b[j],
                        (short)0, acc[i][j], false, false);
        }
        cur ^= 1;   // double buffer: next store goes to the other half
    }

    // ---- epilogue: y = scale[n] * acc + bias[n] ----
#pragma unroll
    for (int j = 0; j < 4; ++j) {
        const int colg = bn + wn * 64 + j * 16 + l16;
        const float s  = scale[colg];
        const float bb = bias[colg];
#pragma unroll
        for (int i = 0; i < 2; ++i) {
#pragma unroll
            for (int v = 0; v < 8; ++v) {
                const int rowg = bm + wm * 32 + i * 16 + v + 8 * g;
                Y[(size_t)rowg * N + colg] = fmaf(s, acc[i][j][v], bb);
            }
        }
    }
}

extern "C" void kernel_launch(void* const* d_in, const int* in_sizes, int n_in,
                              void* d_out, int out_size, void* d_ws, size_t ws_size,
                              hipStream_t stream) {
    const float* X     = (const float*)d_in[0];   // [B,S,D_IN] f32
    const int*   Wq    = (const int*)  d_in[1];   // [D_OUT,D_IN] i32 in [-128,127]
    const float* scale = (const float*)d_in[2];   // [D_OUT]
    const float* bias  = (const float*)d_in[3];   // [D_OUT]
    float* Y = (float*)d_out;

    const int D_OUT = in_sizes[2];            // 4096
    const int D_IN  = in_sizes[1] / D_OUT;    // 4096
    const int M     = in_sizes[0] / D_IN;     // B*S = 8192
    const int N     = D_OUT;
    const int K     = D_IN;

    dim3 grid(N / TN, M / TM);   // 32 x 64 blocks
    dim3 block(256);
    QuantLinear_wmma_kernel<<<grid, block, 0, stream>>>(X, Wq, scale, bias, Y, M, N, K);
}
